// CausalSelfAttention_16587163697313
// MI455X (gfx1250) — matrix operand order repaired and checked by reading, not compiled, not run on hardware
//
#include <hip/hip_runtime.h>

// ---------------------------------------------------------------------------
// Types / helpers
// ---------------------------------------------------------------------------
typedef __bf16  bf16;
typedef __bf16  v16bf __attribute__((ext_vector_type(16)));
typedef float   v8f   __attribute__((ext_vector_type(8)));

union frag16 { v16bf v; uint4 q[2]; };
union bfu4   { uint4 q; bf16 e[8]; };

__device__ __forceinline__ bf16 f2bf(float f) {
  unsigned u = __builtin_bit_cast(unsigned, f);
  unsigned r = (u + 0x7FFFu + ((u >> 16) & 1u)) >> 16;
  unsigned short s = (unsigned short)r;
  return __builtin_bit_cast(bf16, s);
}
__device__ __forceinline__ float bf2f(bf16 b) {
  unsigned short s = __builtin_bit_cast(unsigned short, b);
  unsigned u = ((unsigned)s) << 16;
  return __builtin_bit_cast(float, u);
}

// Async global -> LDS copy (CDNA5 path, bypasses VGPRs; tracked by ASYNCcnt).
// LDS address = low 32 bits of the flat shared pointer.
__device__ __forceinline__ void g2l_b128(void* lds, const void* g) {
  unsigned l = (unsigned)(size_t)lds;
  asm volatile("global_load_async_to_lds_b128 %0, %1, off"
               :: "v"(l), "v"(g)
               : "memory");
}
__device__ __forceinline__ void wait_async0() {
  asm volatile("s_wait_asynccnt 0x0" ::: "memory");
}

#define DIM  2048
#define NH   16
#define HD   128
#define TSEQ 4096
#define EPS  1.1920929e-07f
#define ATT_SCALE 0.08838834764831843f  // 1/sqrt(128)

// ---------------------------------------------------------------------------
// fp32 -> bf16 conversion
// ---------------------------------------------------------------------------
__global__ void cvt_f32_bf16(const float* __restrict__ src, bf16* __restrict__ dst, long n) {
  long i = (long)blockIdx.x * blockDim.x + threadIdx.x;
  if (i < n) dst[i] = f2bf(src[i]);
}

// ---------------------------------------------------------------------------
// v = (1-lamb)*v + lamb*vi   (in-place on bf16 v)
// ---------------------------------------------------------------------------
__global__ void mix_v(bf16* __restrict__ vb, const float* __restrict__ vi,
                      const float* __restrict__ lambp, long n) {
  long i = (long)blockIdx.x * blockDim.x + threadIdx.x;
  if (i < n) {
    float lam = *lambp;
    vb[i] = f2bf((1.0f - lam) * bf2f(vb[i]) + lam * vi[i]);
  }
}

// ---------------------------------------------------------------------------
// GEMM: C[M,N] = A[M,K] * B[N,K]^T  (A,B bf16 row-major, K contiguous)
// block = 256 threads (8 waves); block tile 128(M) x 256(N); wave tile 64x64
// (4x4 WMMA frags, 16 wmma per k-step). Double-buffered LDS, async global->LDS
// staging overlapped with WMMA on the other buffer.
// out_f32 = 1 -> write float, else bf16
// ---------------------------------------------------------------------------
__global__ __launch_bounds__(256) void gemm_bf16_nt(
    const bf16* __restrict__ A, const bf16* __restrict__ B,
    void* __restrict__ Cout, int M, int N, int K, int out_f32)
{
  __shared__ __align__(16) bf16 As[2][128][40];   // 20 KB
  __shared__ __align__(16) bf16 Bs[2][256][40];   // 40 KB

  const int tid  = threadIdx.x;
  const int lane = tid & 31;
  const int wave = tid >> 5;
  const int wm   = wave & 1;   // 2 waves along M
  const int wn   = wave >> 1;  // 4 waves along N
  const int l16  = lane & 15;
  const int lhi  = lane >> 4;

  const int m0 = blockIdx.y * 128;
  const int n0 = blockIdx.x * 256;

  v8f acc[4][4];
#pragma unroll
  for (int i = 0; i < 4; ++i)
#pragma unroll
    for (int j = 0; j < 4; ++j) acc[i][j] = (v8f)0.0f;

  auto stage = [&](int buf, int k0) {
    // A tile: 128x32 = 512 b128 packets (2/thread)
#pragma unroll
    for (int it = 0; it < 2; ++it) {
      int idx = tid * 2 + it;
      int r = idx >> 2;              // 4 packets per 32-wide row
      int c = (idx & 3) * 8;
      g2l_b128(&As[buf][r][c], &A[(size_t)(m0 + r) * K + k0 + c]);
    }
    // B tile: 256x32 = 1024 b128 packets (4/thread)
#pragma unroll
    for (int it = 0; it < 4; ++it) {
      int idx = tid * 4 + it;
      int r = idx >> 2;
      int c = (idx & 3) * 8;
      g2l_b128(&Bs[buf][r][c], &B[(size_t)(n0 + r) * K + k0 + c]);
    }
  };

  const int nk = K >> 5;
  stage(0, 0);
  wait_async0();
  __syncthreads();

  for (int kk = 0; kk < nk; ++kk) {
    const int cur = kk & 1;
    if (kk + 1 < nk) stage(cur ^ 1, (kk + 1) << 5);   // prefetch next tile

    frag16 af[4], bf[4];
#pragma unroll
    for (int i = 0; i < 4; ++i) {
      const bf16* p = &As[cur][wm * 64 + i * 16 + l16][lhi * 8];
      af[i].q[0] = *(const uint4*)p;
      af[i].q[1] = *(const uint4*)(p + 16);
    }
#pragma unroll
    for (int j = 0; j < 4; ++j) {
      const bf16* p = &Bs[cur][wn * 64 + j * 16 + l16][lhi * 8];
      bf[j].q[0] = *(const uint4*)p;
      bf[j].q[1] = *(const uint4*)(p + 16);
    }
#pragma unroll
    for (int i = 0; i < 4; ++i)
#pragma unroll
      for (int j = 0; j < 4; ++j)
        acc[i][j] = __builtin_amdgcn_wmma_f32_16x16x32_bf16(
            false, af[i].v, false, bf[j].v, (short)0, acc[i][j], false, false);

    wait_async0();     // prefetch of next buffer complete
    __syncthreads();   // all waves done reading cur / writes visible
  }

  // store C
#pragma unroll
  for (int i = 0; i < 4; ++i)
#pragma unroll
    for (int j = 0; j < 4; ++j)
#pragma unroll
      for (int r = 0; r < 8; ++r) {
        int row = m0 + wm * 64 + i * 16 + r + lhi * 8;
        int col = n0 + wn * 64 + j * 16 + l16;
        float val = acc[i][j][r];
        if (out_f32) ((float*)Cout)[(size_t)row * N + col] = val;
        else         ((bf16*)Cout)[(size_t)row * N + col]  = f2bf(val);
      }
}

// ---------------------------------------------------------------------------
// Fused RMSNorm + RoPE on one (t, head) row of 128 elems, one wave per row.
// block = (32,8) -> 8 rows per block
// ---------------------------------------------------------------------------
__global__ __launch_bounds__(256) void rms_rope(bf16* __restrict__ buf) {
  int lane  = threadIdx.x;
  int wave  = threadIdx.y;
  int rowid = blockIdx.x * 8 + wave;    // 0 .. T*NH-1
  int t = rowid >> 4;
  int h = rowid & 15;
  bf16* p = buf + (size_t)t * DIM + h * HD;

  float f0 = bf2f(p[lane]);
  float f1 = bf2f(p[lane + 32]);
  float f2 = bf2f(p[lane + 64]);
  float f3 = bf2f(p[lane + 96]);

  float ss = f0 * f0 + f1 * f1 + f2 * f2 + f3 * f3;
#pragma unroll
  for (int off = 1; off < 32; off <<= 1) ss += __shfl_xor(ss, off, 32);
  float rn = rsqrtf(ss * (1.0f / 128.0f) + EPS);

  // rope pairs: (d, d+64) with freq index d (d<64). lane -> d=lane, lane+32.
  float inv0 = __powf(10000.0f, -(float)lane        * (1.0f / 64.0f));
  float inv1 = __powf(10000.0f, -(float)(lane + 32) * (1.0f / 64.0f));
  float th0 = (float)t * inv0, th1 = (float)t * inv1;
  float c0 = __cosf(th0), s0 = __sinf(th0);
  float c1 = __cosf(th1), s1 = __sinf(th1);

  float x1 = f0 * rn, x2 = f2 * rn;
  p[lane]      = f2bf( x1 * c0 + x2 * s0);
  p[lane + 64] = f2bf(-x1 * s0 + x2 * c0);
  x1 = f1 * rn; x2 = f3 * rn;
  p[lane + 32] = f2bf( x1 * c1 + x2 * s1);
  p[lane + 96] = f2bf(-x1 * s1 + x2 * c1);
}

// ---------------------------------------------------------------------------
// Flash-style causal attention, bf16 in / bf16 out.
// grid = (T/128, NH); block = 256 (8 waves); wave w -> q rows [qblk+16w, +16)
// K tile staged via async global->LDS; V tile transposed manually at staging.
// ---------------------------------------------------------------------------
__global__ __launch_bounds__(256) void flash_attn(
    const bf16* __restrict__ Q, const bf16* __restrict__ Km,
    const bf16* __restrict__ V, bf16* __restrict__ Y)
{
  __shared__ __align__(16) bf16 Kt[32][136];     // [kv_local][d]
  __shared__ __align__(16) bf16 Vt[128][40];     // [d][kv_local] (transposed)
  __shared__ __align__(16) bf16 Pb[8][16][40];   // per-wave P tile 16x32

  const int tid  = threadIdx.x;
  const int lane = tid & 31;
  const int wave = tid >> 5;
  const int l16  = lane & 15;
  const int lhi  = lane >> 4;

  const int h     = blockIdx.y;
  const int qblk  = blockIdx.x * 128;
  const int qbase = qblk + wave * 16;
  const size_t hoff = (size_t)h * HD;

  // Q fragments: 4 chunks of K=32 along head dim (kept in VGPRs)
  frag16 aq[4];
  {
    const bf16* qrow = Q + (size_t)(qbase + l16) * DIM + hoff;
#pragma unroll
    for (int c = 0; c < 4; ++c) {
      const bf16* p = qrow + c * 32 + lhi * 8;
      aq[c].q[0] = *(const uint4*)p;
      aq[c].q[1] = *(const uint4*)(p + 16);
    }
  }

  v8f acc[8];
#pragma unroll
  for (int n = 0; n < 8; ++n) acc[n] = (v8f)0.0f;
  float mrow[8], lrow[8];
#pragma unroll
  for (int r = 0; r < 8; ++r) { mrow[r] = -1e30f; lrow[r] = 0.0f; }

  const int kv_end = qblk + 128;  // block causal extent (exclusive)
  for (int kv = 0; kv < kv_end; kv += 32) {
    // ---- stage K tile (async, row-major) and V tile (transposed) ----
#pragma unroll
    for (int it = 0; it < 2; ++it) {
      int idx = tid * 2 + it;          // 0..511
      int r = idx >> 4;                // 16 b128 packets per 128-wide row
      int c = (idx & 15) * 8;
      g2l_b128(&Kt[r][c], &Km[(size_t)(kv + r) * DIM + hoff + c]);
      bfu4 u; u.q = *(const uint4*)&V[(size_t)(kv + r) * DIM + hoff + c];
#pragma unroll
      for (int e = 0; e < 8; ++e) Vt[c + e][r] = u.e[e];
    }
    wait_async0();
    __syncthreads();

    if (kv <= qbase + 15) {
      // ---- S = Q * K^T  (16 q-rows x 32 kv-cols) ----
      v8f s0 = (v8f)0.0f, s1 = (v8f)0.0f;
#pragma unroll
      for (int c = 0; c < 4; ++c) {
        frag16 bk0, bk1;
        const bf16* p0 = &Kt[l16][c * 32 + lhi * 8];
        const bf16* p1 = &Kt[16 + l16][c * 32 + lhi * 8];
        bk0.q[0] = *(const uint4*)p0; bk0.q[1] = *(const uint4*)(p0 + 16);
        bk1.q[0] = *(const uint4*)p1; bk1.q[1] = *(const uint4*)(p1 + 16);
        s0 = __builtin_amdgcn_wmma_f32_16x16x32_bf16(false, aq[c].v, false, bk0.v,
                                                     (short)0, s0, false, false);
        s1 = __builtin_amdgcn_wmma_f32_16x16x32_bf16(false, aq[c].v, false, bk1.v,
                                                     (short)0, s1, false, false);
      }

      // ---- online softmax over the 32 columns ----
#pragma unroll
      for (int r = 0; r < 8; ++r) {
        int row = qbase + r + lhi * 8;
        int c0  = kv + l16;
        int c1  = kv + 16 + l16;
        float v0 = s0[r] * ATT_SCALE; if (c0 > row) v0 = -1e30f;
        float v1 = s1[r] * ATT_SCALE; if (c1 > row) v1 = -1e30f;
        float mx = fmaxf(v0, v1);
#pragma unroll
        for (int off = 1; off < 16; off <<= 1) mx = fmaxf(mx, __shfl_xor(mx, off, 32));
        float mn   = fmaxf(mrow[r], mx);
        float p0   = __expf(v0 - mn);
        float p1   = __expf(v1 - mn);
        float corr = __expf(mrow[r] - mn);
        float rs = p0 + p1;
#pragma unroll
        for (int off = 1; off < 16; off <<= 1) rs += __shfl_xor(rs, off, 32);
        lrow[r] = lrow[r] * corr + rs;
        mrow[r] = mn;
#pragma unroll
        for (int n = 0; n < 8; ++n) acc[n][r] *= corr;
        int lr = r + lhi * 8;
        Pb[wave][lr][l16]      = f2bf(p0);
        Pb[wave][lr][16 + l16] = f2bf(p1);
      }

      // ---- acc += P * V  ----
      frag16 ap;
      {
        const bf16* p = &Pb[wave][l16][lhi * 8];
        ap.q[0] = *(const uint4*)p;
        ap.q[1] = *(const uint4*)(p + 16);
      }
#pragma unroll
      for (int n = 0; n < 8; ++n) {
        frag16 bv;
        const bf16* vp = &Vt[n * 16 + l16][lhi * 8];
        bv.q[0] = *(const uint4*)vp;
        bv.q[1] = *(const uint4*)(vp + 16);
        acc[n] = __builtin_amdgcn_wmma_f32_16x16x32_bf16(false, ap.v, false, bv.v,
                                                         (short)0, acc[n], false, false);
      }
    }
    __syncthreads();
  }

  // ---- finalize: y = acc / l ----
#pragma unroll
  for (int n = 0; n < 8; ++n) {
#pragma unroll
    for (int r = 0; r < 8; ++r) {
      int row = qbase + r + lhi * 8;
      int col = n * 16 + l16;
      float val = acc[n][r] / lrow[r];
      Y[(size_t)row * DIM + hoff + col] = f2bf(val);
    }
  }
}

// ---------------------------------------------------------------------------
// Launch
// ---------------------------------------------------------------------------
extern "C" void kernel_launch(void* const* d_in, const int* in_sizes, int n_in,
                              void* d_out, int out_size, void* d_ws, size_t ws_size,
                              hipStream_t stream) {
  const float* x    = (const float*)d_in[0];
  const float* vi   = (const float*)d_in[1];
  const float* Wq   = (const float*)d_in[2];
  const float* Wk   = (const float*)d_in[3];
  const float* Wv   = (const float*)d_in[4];
  const float* Wp   = (const float*)d_in[5];
  const float* lamb = (const float*)d_in[6];
  float* out = (float*)d_out;

  char* ws = (char*)d_ws;
  const size_t MB = 1024ull * 1024ull;
  bf16* XB  = (bf16*)(ws + 0 * MB);    // 4096x2048 bf16 : 16 MB
  bf16* WQB = (bf16*)(ws + 16 * MB);   // 2048x2048 bf16 :  8 MB
  bf16* WKB = (bf16*)(ws + 24 * MB);
  bf16* WVB = (bf16*)(ws + 32 * MB);
  bf16* WPB = (bf16*)(ws + 40 * MB);
  bf16* QB  = (bf16*)(ws + 48 * MB);   // 16 MB each
  bf16* KB  = (bf16*)(ws + 64 * MB);
  bf16* VB  = (bf16*)(ws + 80 * MB);
  bf16* YB  = (bf16*)(ws + 96 * MB);

  const long nXD = (long)TSEQ * DIM;   // 8M
  const long nWW = (long)DIM * DIM;    // 4M

  // 1) convert operands to bf16 (L2-resident afterwards: total ~112MB < 192MB L2)
  cvt_f32_bf16<<<(int)(nXD / 256), 256, 0, stream>>>(x,  XB,  nXD);
  cvt_f32_bf16<<<(int)(nWW / 256), 256, 0, stream>>>(Wq, WQB, nWW);
  cvt_f32_bf16<<<(int)(nWW / 256), 256, 0, stream>>>(Wk, WKB, nWW);
  cvt_f32_bf16<<<(int)(nWW / 256), 256, 0, stream>>>(Wv, WVB, nWW);
  cvt_f32_bf16<<<(int)(nWW / 256), 256, 0, stream>>>(Wp, WPB, nWW);

  // 2) QKV projections (WMMA bf16, bf16 out)
  dim3 gg(DIM / 256, TSEQ / 128);  // (8, 32)
  gemm_bf16_nt<<<gg, 256, 0, stream>>>(XB, WQB, QB, TSEQ, DIM, DIM, 0);
  gemm_bf16_nt<<<gg, 256, 0, stream>>>(XB, WKB, KB, TSEQ, DIM, DIM, 0);
  gemm_bf16_nt<<<gg, 256, 0, stream>>>(XB, WVB, VB, TSEQ, DIM, DIM, 0);

  // 3) v mix + q/k rmsnorm+rope
  mix_v<<<(int)(nXD / 256), 256, 0, stream>>>(VB, vi, lamb, nXD);
  rms_rope<<<TSEQ * NH / 8, dim3(32, 8), 0, stream>>>(QB);
  rms_rope<<<TSEQ * NH / 8, dim3(32, 8), 0, stream>>>(KB);

  // 4) flash causal attention
  flash_attn<<<dim3(TSEQ / 128, NH), 256, 0, stream>>>(QB, KB, VB, YB);

  // 5) output projection -> fp32 d_out
  gemm_bf16_nt<<<gg, 256, 0, stream>>>(YB, WPB, out, TSEQ, DIM, DIM, 1);
}
